// CustomRobertaAttention_2551210574124
// MI455X (gfx1250) — matrix operand order repaired and checked by reading, not compiled, not run on hardware
//
#include <hip/hip_runtime.h>
#include <hip/hip_bf16.h>

// ---------------------------------------------------------------------------
// RoBERTa attention block for MI455X (gfx1250).
//   bf16 WMMA + flash attention + async LDS staging (ASYNCcnt).
//   H=1024, NH=16, HD=64, B=4, S=2048
// ---------------------------------------------------------------------------

#define HDIM 1024
#define SEQ  2048
#define BATCH 4
#define NHEAD 16
#define HD 64

typedef __bf16 bf16_t;
typedef __attribute__((ext_vector_type(16))) __bf16 v16bf;
typedef __attribute__((ext_vector_type(8)))  float  v8f;
typedef __attribute__((ext_vector_type(4)))  unsigned int uv4;

union AFrag { v16bf v; uv4 q[2]; };
union Pack8 { uv4 q; __bf16 h[8]; };

__device__ __forceinline__ v8f wmma_bf16(v16bf a, v16bf b, v8f c) {
#if defined(__HIP_DEVICE_COMPILE__)
    // D = A(16x32 bf16) * B(32x16 bf16) + C(16x16 f32)
    return __builtin_amdgcn_wmma_f32_16x16x32_bf16(false, a, false, b,
                                                   (short)0, c, false, false);
#else
    (void)a; (void)b;
    return c;
#endif
}

__device__ __forceinline__ uv4 ld128(const bf16_t* p) { return *(const uv4*)p; }

// 4KB linear async copy global->LDS: 8 x (32 lanes x 16B) b128 transfers.
__device__ __forceinline__ void async_copy_4k(unsigned lds_base, const char* gsrc, int lane) {
#if defined(__HIP_DEVICE_COMPILE__)
    unsigned l = lds_base + lane * 16;
    const char* g = gsrc + lane * 16;
#pragma unroll
    for (int i = 0; i < 8; ++i) {
        asm volatile("global_load_async_to_lds_b128 %0, %1, off"
                     :: "v"(l + i * 512), "v"(g + i * 512) : "memory");
    }
#endif
}
__device__ __forceinline__ void wait_async_le16() {
#if defined(__HIP_DEVICE_COMPILE__)
    asm volatile("s_wait_asynccnt 0x10" ::: "memory");   // prev chunk done, next in flight
#endif
}
__device__ __forceinline__ void wait_async_0() {
#if defined(__HIP_DEVICE_COMPILE__)
    asm volatile("s_wait_asynccnt 0x0" ::: "memory");
#endif
}

// ---------------------------------------------------------------------------
// Prep: fp32 -> bf16 conversion of hidden states (row-major [8192][1024])
// ---------------------------------------------------------------------------
__global__ void k_cvt_x(const float* __restrict__ x, bf16_t* __restrict__ o, int n4) {
    int i = blockIdx.x * 256 + threadIdx.x;
    if (i >= n4) return;
    float4 v = ((const float4*)x)[i];
    union { unsigned long long u; __bf16 h[4]; } p;
    p.h[0] = (bf16_t)v.x; p.h[1] = (bf16_t)v.y;
    p.h[2] = (bf16_t)v.z; p.h[3] = (bf16_t)v.w;
    ((unsigned long long*)o)[i] = p.u;
}

// ---------------------------------------------------------------------------
// Prep: weight transpose + convert: Wt[n][k] = (bf16) W[k][n]   (1024x1024)
// ---------------------------------------------------------------------------
__global__ void k_cvt_wt(const float* __restrict__ w, bf16_t* __restrict__ wt) {
    const int n   = blockIdx.x;
    const int tid = threadIdx.x;
    const int k0  = tid * 4;
    union { unsigned long long u; __bf16 h[4]; } p;
    for (int j = 0; j < 4; ++j)
        p.h[j] = (bf16_t)w[(size_t)(k0 + j) * HDIM + n];
    ((unsigned long long*)(wt + (size_t)n * HDIM))[tid] = p.u;
}

// ---------------------------------------------------------------------------
// Software-pipelined 32x64 GEMM fragments
// ---------------------------------------------------------------------------
struct GFrags { AFrag a0, a1, b[4]; };

__device__ __forceinline__ void gemm_load(GFrags& f, const bf16_t* Xr0, const bf16_t* Xr1,
                                          const bf16_t* Wl, int k, int lh) {
    f.a0.q[0] = ld128(Xr0 + k + lh * 8);
    f.a0.q[1] = ld128(Xr0 + k + 16 + lh * 8);
    f.a1.q[0] = ld128(Xr1 + k + lh * 8);
    f.a1.q[1] = ld128(Xr1 + k + 16 + lh * 8);
#pragma unroll
    for (int t = 0; t < 4; ++t) {
        const bf16_t* Wc = Wl + (size_t)t * 16 * HDIM + k + lh * 8;
        f.b[t].q[0] = ld128(Wc);
        f.b[t].q[1] = ld128(Wc + 16);
    }
}
__device__ __forceinline__ void gemm_mma(v8f acc[2][4], const GFrags& f) {
#pragma unroll
    for (int t = 0; t < 4; ++t) {
        acc[0][t] = wmma_bf16(f.a0.v, f.b[t].v, acc[0][t]);
        acc[1][t] = wmma_bf16(f.a1.v, f.b[t].v, acc[1][t]);
    }
}

// ---------------------------------------------------------------------------
// Kernel 1: fused QKV projection GEMM (bf16 WMMA, f32 acc, pipelined).
//   X [8192 x 1024] bf16; Wt [n][k] bf16.
//   Q,K -> [B,NH,S,HD]; V -> chunked [B*NH][S/32][64][32] (4KB tiles for async)
// grid = (256 m-tiles of 32 rows, 48 = 3 mats x 16 heads), block = 32
// ---------------------------------------------------------------------------
__global__ void __launch_bounds__(32)
k_qkv_gemm(const bf16_t* __restrict__ X,
           const bf16_t* __restrict__ Wqt, const bf16_t* __restrict__ Wkt,
           const bf16_t* __restrict__ Wvt,
           const float* __restrict__ bq, const float* __restrict__ bk,
           const float* __restrict__ bv,
           bf16_t* __restrict__ Qb, bf16_t* __restrict__ Kb,
           bf16_t* __restrict__ Vcb) {
    const int mt  = blockIdx.x;        // 32 rows each
    const int g   = blockIdx.y;
    const int mat = g >> 4;
    const int h   = g & 15;
    const bf16_t* W    = (mat == 0) ? Wqt : ((mat == 1) ? Wkt : Wvt);
    const float*  bias = (mat == 0) ? bq  : ((mat == 1) ? bk  : bv);

    const int lane = threadIdx.x, l15 = lane & 15, lh = lane >> 4;
    const bf16_t* Xr0 = X + (size_t)(mt * 32 + l15) * HDIM;
    const bf16_t* Xr1 = Xr0 + (size_t)16 * HDIM;
    const bf16_t* Wl  = W + (size_t)(h * HD + l15) * HDIM;

    v8f acc[2][4];
    const v8f vzero = {0.f,0.f,0.f,0.f,0.f,0.f,0.f,0.f};
    for (int i = 0; i < 2; ++i)
        for (int t = 0; t < 4; ++t) acc[i][t] = vzero;

    GFrags f0, f1;
    gemm_load(f0, Xr0, Xr1, Wl, 0, lh);
    for (int k = 0; k < HDIM - 64; k += 64) {
        gemm_load(f1, Xr0, Xr1, Wl, k + 32, lh);
        __builtin_prefetch(Xr0 + k + 128, 0, 0);
        gemm_mma(acc, f0);
        gemm_load(f0, Xr0, Xr1, Wl, k + 64, lh);
        gemm_mma(acc, f1);
    }
    gemm_load(f1, Xr0, Xr1, Wl, HDIM - 32, lh);
    gemm_mma(acc, f0);
    gemm_mma(acc, f1);

    const int bidx = mt >> 6;                         // batch
    const int s0   = (mt & 63) * 32;                  // seq base (32-aligned)
    const int chnk = s0 >> 5;                         // V chunk index
    const size_t hb = (size_t)bidx * NHEAD + h;
#pragma unroll
    for (int h2 = 0; h2 < 2; ++h2) {
#pragma unroll
        for (int t = 0; t < 4; ++t) {
            const int d = t * 16 + l15;
            const float bsv = bias[h * HD + d];
            if (mat < 2) {
                bf16_t* O = ((mat == 0) ? Qb : Kb) + hb * (size_t)SEQ * HD;
                for (int j = 0; j < 8; ++j) {
                    const int s = s0 + h2 * 16 + j + lh * 8;
                    O[(size_t)s * HD + d] = (bf16_t)(acc[h2][t][j] + bsv);
                }
            } else {
                Pack8 pk;
                for (int j = 0; j < 8; ++j) pk.h[j] = (bf16_t)(acc[h2][t][j] + bsv);
                *(uv4*)(Vcb + (((size_t)hb * (SEQ / 32) + chnk) * 64 + d) * 32
                        + h2 * 16 + lh * 8) = pk.q;
            }
        }
    }
}

// ---------------------------------------------------------------------------
// Kernel 2: flash attention. One wave per (batch*head, 16-query tile).
//   S^T = K*Q^T (softmax stats per-lane), ctx^T = V^T * P^T.
//   K/V tiles double-buffered in LDS via global_load_async_to_lds_b128.
// grid = (128 q-tiles, 64 batch-heads), block = 32
// ---------------------------------------------------------------------------
__global__ void __launch_bounds__(32)
k_attn(const bf16_t* __restrict__ Qb, const bf16_t* __restrict__ Kb,
       const bf16_t* __restrict__ Vcb, const float* __restrict__ mask,
       bf16_t* __restrict__ Ctx) {
    __shared__ __align__(16) bf16_t shK[2][32 * HD];   // 2 x 4KB, [key][d]
    __shared__ __align__(16) bf16_t shV[2][HD * 32];   // 2 x 4KB, [d][key]

    const int qt   = blockIdx.x;
    const int bh   = blockIdx.y;
    const int lane = threadIdx.x, l15 = lane & 15, lh = lane >> 4;
    const int bidx = bh >> 4;
    const int hidx = bh & 15;

    // LDS byte offsets (flat shared addr low 32 bits = LDS offset; aperture is [63:32])
    const unsigned ldsK = (unsigned)(size_t)&shK[0][0];
    const unsigned ldsV = (unsigned)(size_t)&shV[0][0];
    const char* Kbase = (const char*)(Kb + (size_t)bh * SEQ * HD);               // 4KB/chunk
    const char* Vbase = (const char*)(Vcb + (size_t)bh * (SEQ / 32) * (64 * 32)); // 4KB/chunk

    // Q fragments (B operand of S^T) stay in registers
    const bf16_t* Qrow = Qb + ((size_t)bh * SEQ + qt * 16 + l15) * HD;
    AFrag qf0, qf1;
    qf0.q[0] = ld128(Qrow + lh * 8);
    qf0.q[1] = ld128(Qrow + 16 + lh * 8);
    qf1.q[0] = ld128(Qrow + 32 + lh * 8);
    qf1.q[1] = ld128(Qrow + 48 + lh * 8);

    v8f acc[4];
    const v8f vzero = {0.f,0.f,0.f,0.f,0.f,0.f,0.f,0.f};
    for (int t = 0; t < 4; ++t) acc[t] = vzero;

    float mrun = -3.0e38f, lrun = 0.f;
    const float* mrow = mask + (size_t)bidx * SEQ;

    // prefetch chunk 0
    async_copy_4k(ldsK, Kbase, lane);
    async_copy_4k(ldsV, Vbase, lane);

    for (int c = 0; c < SEQ / 32; ++c) {
        const int p  = c & 1;
        const int kv = c * 32;
        if (c + 1 < SEQ / 32) {
            const int pn = (c + 1) & 1;
            async_copy_4k(ldsK + pn * 4096, Kbase + (size_t)(c + 1) * 4096, lane);
            async_copy_4k(ldsV + pn * 4096, Vbase + (size_t)(c + 1) * 4096, lane);
            wait_async_le16();            // chunk c resident, chunk c+1 in flight
        } else {
            wait_async_0();
        }

        // ---- S^T tiles from LDS (rows = keys, cols = queries) ----
        const bf16_t* kr0 = &shK[p][l15 * HD];
        const bf16_t* kr1 = &shK[p][(16 + l15) * HD];
        AFrag ka;
        v8f s0 = vzero, s1 = vzero;
        ka.q[0] = ld128(kr0 + lh * 8);      ka.q[1] = ld128(kr0 + 16 + lh * 8);
        s0 = wmma_bf16(ka.v, qf0.v, s0);
        ka.q[0] = ld128(kr0 + 32 + lh * 8); ka.q[1] = ld128(kr0 + 48 + lh * 8);
        s0 = wmma_bf16(ka.v, qf1.v, s0);
        ka.q[0] = ld128(kr1 + lh * 8);      ka.q[1] = ld128(kr1 + 16 + lh * 8);
        s1 = wmma_bf16(ka.v, qf0.v, s1);
        ka.q[0] = ld128(kr1 + 32 + lh * 8); ka.q[1] = ld128(kr1 + 48 + lh * 8);
        s1 = wmma_bf16(ka.v, qf1.v, s1);

        // additive mask indexed by key position (vgpr row)
        const float4 m0a = *(const float4*)(mrow + kv + lh * 8);
        const float4 m0b = *(const float4*)(mrow + kv + lh * 8 + 4);
        const float4 m1a = *(const float4*)(mrow + kv + 16 + lh * 8);
        const float4 m1b = *(const float4*)(mrow + kv + 16 + lh * 8 + 4);
        const float mk0[8] = {m0a.x,m0a.y,m0a.z,m0a.w,m0b.x,m0b.y,m0b.z,m0b.w};
        const float mk1[8] = {m1a.x,m1a.y,m1a.z,m1a.w,m1b.x,m1b.y,m1b.z,m1b.w};

        // ---- online softmax (per-lane; combine halves via shfl_xor 16) ----
        float mloc = -3.0e38f;
        for (int j = 0; j < 8; ++j) {
            s0[j] = s0[j] * 0.125f + mk0[j];
            s1[j] = s1[j] * 0.125f + mk1[j];
            mloc = fmaxf(mloc, fmaxf(s0[j], s1[j]));
        }
        mloc = fmaxf(mloc, __shfl_xor(mloc, 16));
        const float mnew = fmaxf(mrun, mloc);
        const float resc = __expf(mrun - mnew);
        float p0[8], p1[8], psum = 0.f;
        for (int j = 0; j < 8; ++j) {
            p0[j] = __expf(s0[j] - mnew);
            p1[j] = __expf(s1[j] - mnew);
            psum += p0[j] + p1[j];
        }
        psum += __shfl_xor(psum, 16);
        lrun = lrun * resc + psum;
        mrun = mnew;
        for (int t = 0; t < 4; ++t)
            for (int j = 0; j < 8; ++j) acc[t][j] *= resc;

        // ---- build P^T B-fragment; swap missing key subset across halves ----
        union { v16bf v; __bf16 h[16]; } pf;
        for (int j = 0; j < 8; ++j) {
            const float o0 = __shfl_xor(p0[j], 16);
            const float o1 = __shfl_xor(p1[j], 16);
            pf.h[j]     = (bf16_t)((lh == 0) ? p0[j] : o1);
            pf.h[8 + j] = (bf16_t)((lh == 0) ? o0    : p1[j]);
        }

        // ---- ctx^T accumulate: A = rows of V^T (d) from LDS ----
#pragma unroll
        for (int t = 0; t < 4; ++t) {
            const bf16_t* vr = &shV[p][(t * 16 + l15) * 32];
            AFrag va;
            va.q[0] = ld128(vr + lh * 16);
            va.q[1] = ld128(vr + lh * 16 + 8);
            acc[t] = wmma_bf16(va.v, pf.v, acc[t]);
        }
    }

    const float invl = 1.0f / lrun;
    const size_t mrow_out = (size_t)bidx * SEQ + qt * 16 + l15;
    for (int t = 0; t < 4; ++t) {
        Pack8 pk;
        for (int j = 0; j < 8; ++j) pk.h[j] = (bf16_t)(acc[t][j] * invl);
        *(uv4*)(Ctx + mrow_out * HDIM + hidx * HD + t * 16 + lh * 8) = pk.q;
    }
}

// ---------------------------------------------------------------------------
// Kernel 3: output projection Y = Ctx @ Wo + bo + residual (f32, pipelined)
// grid = (256 m-tiles of 32 rows, 16 n-groups of 64), block = 32
// ---------------------------------------------------------------------------
__global__ void __launch_bounds__(32)
k_out_gemm(const bf16_t* __restrict__ C, const bf16_t* __restrict__ Wot,
           const float* __restrict__ bo, const float* __restrict__ resid,
           float* __restrict__ Y) {
    const int mt = blockIdx.x;
    const int ng = blockIdx.y;
    const int lane = threadIdx.x, l15 = lane & 15, lh = lane >> 4;
    const bf16_t* Cr0 = C + (size_t)(mt * 32 + l15) * HDIM;
    const bf16_t* Cr1 = Cr0 + (size_t)16 * HDIM;
    const bf16_t* Wl  = Wot + (size_t)(ng * 64 + l15) * HDIM;

    v8f acc[2][4];
    const v8f vzero = {0.f,0.f,0.f,0.f,0.f,0.f,0.f,0.f};
    for (int i = 0; i < 2; ++i)
        for (int t = 0; t < 4; ++t) acc[i][t] = vzero;

    GFrags f0, f1;
    gemm_load(f0, Cr0, Cr1, Wl, 0, lh);
    for (int k = 0; k < HDIM - 64; k += 64) {
        gemm_load(f1, Cr0, Cr1, Wl, k + 32, lh);
        __builtin_prefetch(Cr0 + k + 128, 0, 0);
        gemm_mma(acc, f0);
        gemm_load(f0, Cr0, Cr1, Wl, k + 64, lh);
        gemm_mma(acc, f1);
    }
    gemm_load(f1, Cr0, Cr1, Wl, HDIM - 32, lh);
    gemm_mma(acc, f0);
    gemm_mma(acc, f1);

#pragma unroll
    for (int h2 = 0; h2 < 2; ++h2) {
#pragma unroll
        for (int t = 0; t < 4; ++t) {
            const int n = ng * 64 + t * 16 + l15;
            const float bsv = bo[n];
            for (int j = 0; j < 8; ++j) {
                const size_t m = (size_t)mt * 32 + h2 * 16 + j + lh * 8;
                Y[m * HDIM + n] = acc[h2][t][j] + bsv + resid[m * HDIM + n];
            }
        }
    }
}

// ---------------------------------------------------------------------------
// Kernel 4: LayerNorm over 1024 per row. grid = 8192 rows, block = 256.
// ---------------------------------------------------------------------------
__global__ void k_layernorm(const float* __restrict__ Yv,
                            const float* __restrict__ g,
                            const float* __restrict__ b,
                            float* __restrict__ out) {
    const int row = blockIdx.x;
    const int tid = threadIdx.x;
    const float4 v = ((const float4*)(Yv + (size_t)row * HDIM))[tid];
    float s  = v.x + v.y + v.z + v.w;
    float s2 = v.x * v.x + v.y * v.y + v.z * v.z + v.w * v.w;
    for (int off = 16; off > 0; off >>= 1) {
        s  += __shfl_xor(s, off);
        s2 += __shfl_xor(s2, off);
    }
    __shared__ float ls[8], ls2[8];
    const int w = tid >> 5;
    if ((tid & 31) == 0) { ls[w] = s; ls2[w] = s2; }
    __syncthreads();
    float ts = 0.f, ts2 = 0.f;
    for (int i = 0; i < 8; ++i) { ts += ls[i]; ts2 += ls2[i]; }
    const float mu  = ts * (1.0f / HDIM);
    const float var = ts2 * (1.0f / HDIM) - mu * mu;
    const float r   = rsqrtf(var + 1e-12f);
    const float4 gv = ((const float4*)g)[tid];
    const float4 bv = ((const float4*)b)[tid];
    float4 ov;
    ov.x = (v.x - mu) * r * gv.x + bv.x;
    ov.y = (v.y - mu) * r * gv.y + bv.y;
    ov.z = (v.z - mu) * r * gv.z + bv.z;
    ov.w = (v.w - mu) * r * gv.w + bv.w;
    ((float4*)(out + (size_t)row * HDIM))[tid] = ov;
}

// ---------------------------------------------------------------------------
extern "C" void kernel_launch(void* const* d_in, const int* in_sizes, int n_in,
                              void* d_out, int out_size, void* d_ws, size_t ws_size,
                              hipStream_t stream) {
    (void)in_sizes; (void)n_in; (void)out_size; (void)ws_size;
    const float* hs   = (const float*)d_in[0];
    const float* mask = (const float*)d_in[1];
    const float* Wq   = (const float*)d_in[2];
    const float* bq   = (const float*)d_in[3];
    const float* Wk   = (const float*)d_in[4];
    const float* bk   = (const float*)d_in[5];
    const float* Wv   = (const float*)d_in[6];
    const float* bv   = (const float*)d_in[7];
    const float* Wo   = (const float*)d_in[8];
    const float* bo   = (const float*)d_in[9];
    const float* lng  = (const float*)d_in[10];
    const float* lnb  = (const float*)d_in[11];
    float* out = (float*)d_out;

    char* ws = (char*)d_ws;
    const size_t MB = 1ull << 20;
    bf16_t* Xbf = (bf16_t*)(ws + 0);          // 16 MB  [8192][1024] bf16
    bf16_t* Wqt = (bf16_t*)(ws + 16 * MB);    //  2 MB  transposed bf16
    bf16_t* Wkt = (bf16_t*)(ws + 18 * MB);
    bf16_t* Wvt = (bf16_t*)(ws + 20 * MB);
    bf16_t* Wot = (bf16_t*)(ws + 22 * MB);
    bf16_t* Qb  = (bf16_t*)(ws + 24 * MB);    // 16 MB  [B,NH,S,HD]
    bf16_t* Kb  = (bf16_t*)(ws + 40 * MB);    // 16 MB  [B,NH,S,HD]
    bf16_t* Vcb = (bf16_t*)(ws + 56 * MB);    // 16 MB  [B*NH][S/32][64][32]
    bf16_t* Ctx = (bf16_t*)(ws + 72 * MB);    // 16 MB  [8192][1024]
    float*  Yb  = (float*)(ws + 88 * MB);     // 32 MB  pre-LN f32

    k_cvt_x<<<dim3(8192), dim3(256), 0, stream>>>(hs, Xbf, (BATCH * SEQ * HDIM) / 4);
    k_cvt_wt<<<dim3(1024), dim3(256), 0, stream>>>(Wq, Wqt);
    k_cvt_wt<<<dim3(1024), dim3(256), 0, stream>>>(Wk, Wkt);
    k_cvt_wt<<<dim3(1024), dim3(256), 0, stream>>>(Wv, Wvt);
    k_cvt_wt<<<dim3(1024), dim3(256), 0, stream>>>(Wo, Wot);

    k_qkv_gemm<<<dim3(256, 48), dim3(32), 0, stream>>>(Xbf, Wqt, Wkt, Wvt,
                                                       bq, bk, bv, Qb, Kb, Vcb);
    k_attn<<<dim3(128, 64), dim3(32), 0, stream>>>(Qb, Kb, Vcb, mask, Ctx);
    k_out_gemm<<<dim3(256, 16), dim3(32), 0, stream>>>(Ctx, Wot, bo, hs, Yb);
    k_layernorm<<<dim3(8192), dim3(256), 0, stream>>>(Yb, lng, lnb, out);
}
